// SO3Transformer_27487790694416
// MI455X (gfx1250) — matrix operand order repaired and checked by reading, compiled, not run on hardware
//
#include <hip/hip_runtime.h>
#include <math.h>

// ---------------------------------------------------------------------------
// SO3 transformer layer for MI455X (gfx1250, wave32, WMMA).
// Dominant GEMM (h[16x32] @ Rw2[32x128] per pair-tile) runs on
// v_wmma_f32_16x16x32_f16.  B fragments live in LDS; a per-tile scheduling
// fence caps live WMMA D-tiles (keeps VGPRs < 256).  vals tiles are staged
// memory->LDS with GLOBAL_LOAD_ASYNC_TO_LDS_B128 (ASYNCcnt + s_wait_asynccnt);
// a synchronous float4 path remains as fallback for toolchains without it.
// ---------------------------------------------------------------------------

typedef __attribute__((ext_vector_type(16))) _Float16 v16h;
typedef __attribute__((ext_vector_type(8)))  float    v8f;

#if defined(__AMDGCN__) && __has_builtin(__builtin_amdgcn_global_load_async_to_lds_b128) && __has_builtin(__builtin_amdgcn_s_wait_asynccnt)
#define HAVE_ASYNC_LDS 1
typedef int v4i_ __attribute__((vector_size(16)));   // builtin's pointee type
#else
#define HAVE_ASYNC_LDS 0
#endif

#define B_   8
#define N_   256
#define FIN  4
#define C_   32
#define L_   4
#define LC   128      // L*C
#define H_   32
#define FOUT 256
#define DOUT 40
#define EPSF 1e-8f

// workspace layout (float offsets)
#define WS_Q     0
#define WS_K     (WS_Q + B_*N_*C_)          //  65536
#define WS_VALS  (WS_K + B_*N_*C_)          // 131072
#define WS_ALPHA (WS_VALS + B_*N_*LC)       // 393216
#define WS_INV   (WS_ALPHA + B_*N_*N_)      // 917504
// total: 1179648 floats = 4.5 MB scratch (lives in the 192 MB L2)

// ---------------------------------------------------------------------------
// Kernel 1: q = x@Wq, k = x@Wk, vals[b,n,l,c] = x@Wv  (4-MAC dots, VALU)
// ---------------------------------------------------------------------------
__global__ void __launch_bounds__(128)
prep_kernel(const float* __restrict__ x, const float* __restrict__ Wq,
            const float* __restrict__ Wk, const float* __restrict__ Wv,
            float* __restrict__ ws) {
  const int bn = blockIdx.x;          // b*N + n
  const int t  = threadIdx.x;         // 0..127 -> lc
  const float* xp = x + (size_t)bn * FIN;
  const float x0 = xp[0], x1 = xp[1], x2 = xp[2], x3 = xp[3];

  const int l = t >> 5, c = t & 31;
  const float* wv = Wv + (size_t)(l * FIN) * C_ + c;
  ws[WS_VALS + (size_t)bn * LC + t] =
      x0 * wv[0] + x1 * wv[C_] + x2 * wv[2 * C_] + x3 * wv[3 * C_];

  if (t < C_) {
    ws[WS_Q + (size_t)bn * C_ + t] =
        x0 * Wq[t] + x1 * Wq[C_ + t] + x2 * Wq[2 * C_ + t] + x3 * Wq[3 * C_ + t];
  } else if (t < 2 * C_) {
    const int c2 = t - C_;
    ws[WS_K + (size_t)bn * C_ + c2] =
        x0 * Wk[c2] + x1 * Wk[C_ + c2] + x2 * Wk[2 * C_ + c2] + x3 * Wk[3 * C_ + c2];
  }
}

// ---------------------------------------------------------------------------
// Kernel 2: alpha[b,i,:] = softmax_j( q[b,i]·k[b,j] / sqrt(C) )
// ---------------------------------------------------------------------------
__global__ void __launch_bounds__(256)
attn_kernel(float* __restrict__ ws) {
  const int bi = blockIdx.x;          // b*N + i
  const int b  = bi / N_;
  const int j  = threadIdx.x;
  __shared__ float qs[C_];
  __shared__ float red[256];

  if (j < C_) qs[j] = ws[WS_Q + (size_t)bi * C_ + j];
  __syncthreads();

  const float* kp = ws + WS_K + ((size_t)b * N_ + j) * C_;
  float lg = 0.f;
  #pragma unroll
  for (int c = 0; c < C_; ++c) lg += qs[c] * kp[c];
  lg *= 0.17677669529663687f;         // 1/sqrt(32)

  red[j] = lg; __syncthreads();
  for (int s = 128; s > 0; s >>= 1) {
    if (j < s) red[j] = fmaxf(red[j], red[j + s]);
    __syncthreads();
  }
  const float mx = red[0]; __syncthreads();
  const float e = __expf(lg - mx);
  red[j] = e; __syncthreads();
  for (int s = 128; s > 0; s >>= 1) {
    if (j < s) red[j] += red[j + s];
    __syncthreads();
  }
  ws[WS_ALPHA + (size_t)bi * N_ + j] = e / red[0];
}

// ---------------------------------------------------------------------------
// Kernel 3: fused per-(b,i) pair pipeline.  One wave32 per (b,i).
//   per j-tile of 16:  stage vals tile -> geometry -> Y[16] -> A-frag h(f16)
//   -> 8x { WMMA (h @ Rw2 tile, B from LDS) -> einsum accumulate } serialized
// ---------------------------------------------------------------------------
__global__ void __launch_bounds__(32)
pair_kernel(const float* __restrict__ x,
            const float* __restrict__ Rw1, const float* __restrict__ Rb1,
            const float* __restrict__ Rw2, const float* __restrict__ Rb2,
            float* __restrict__ ws) {
  const int bi = blockIdx.x;
  const int b = bi / N_, i = bi % N_;
  const int lane = threadIdx.x;
  const int l0 = lane & 15;           // row (A) / column (B,D) index
  const int hi = lane >> 4;           // half-wave select

  __shared__ float4 Vlds4[16 * LC / 4];   // 8 KB: vals (or alpha*vals) tile
  __shared__ v16h   Bh[8][32];            // 8 KB: Rw2 f16 B fragments, 1 slot/lane
  __shared__ float  Ylds[16 * 16];        // 16 Y components x 16 j
  __shared__ float  Alds[16];             // alpha for the tile (async path)
  float* Vlds = (float*)Vlds4;

  // Radial-MLP constants in packed f16, A-fragment K order:
  //   e<8  -> K = e + 8*hi ; e>=8 -> K = e+8 + 8*hi   (ISA 16-bit A 16x32)
  v16h wh, bh;
  #pragma unroll
  for (int e = 0; e < 16; ++e) {
    const int K = (e < 8 ? e : e + 8) + hi * 8;
    wh[e] = (_Float16)Rw1[K];
    bh[e] = (_Float16)Rb1[K];
  }

  // Stage Rw2 B fragments into LDS (each lane owns its 32-byte slot).
  // B K order matches A: e<8 -> K = e + 8*hi ; e>=8 -> K = e+8 + 8*hi ;
  // column = t*16 + (lane&15).
  float rb2v[8];
  #pragma unroll
  for (int t = 0; t < 8; ++t) {
    const int col = t * 16 + l0;
    rb2v[t] = Rb2[col];
    v16h tmp;
    #pragma unroll
    for (int e = 0; e < 16; ++e)
      tmp[e] = (_Float16)Rw2[((e < 8 ? e : e + 8) + hi * 8) * LC + col];
    Bh[t][lane] = tmp;
  }
  __syncthreads();

  const float cix = x[((size_t)b * N_ + i) * FIN + 0];
  const float ciy = x[((size_t)b * N_ + i) * FIN + 1];
  const float ciz = x[((size_t)b * N_ + i) * FIN + 2];

  float acc[32];                      // per-lane (c,m) accumulators: 2*(1+3+5+7)
  #pragma unroll
  for (int m = 0; m < 32; ++m) acc[m] = 0.f;

  const float* vals_g  = ws + WS_VALS + (size_t)b * N_ * LC;
  const float* alpha_g = ws + WS_ALPHA + (size_t)bi * N_;

  static constexpr int mcount[8] = {1, 1, 3, 3, 5, 5, 7, 7};
  static constexpr int moff[8]   = {0, 0, 1, 1, 4, 4, 9, 9};
  static constexpr int aoff[8]   = {0, 1, 2, 5, 8, 13, 18, 25};

  for (int jt = 0; jt < 16; ++jt) {
    const int j0 = jt * 16;
    __syncthreads();                  // previous tile fully consumed

#if HAVE_ASYNC_LDS
    // Direct memory->LDS DMA: no VGPR data path, tracked by ASYNCcnt.
    // Iteration 'it' stages row j0+it; each lane moves one 16-byte chunk.
    #pragma unroll
    for (int it = 0; it < 16; ++it) {
      __builtin_amdgcn_global_load_async_to_lds_b128(
          (v4i_*)(vals_g + (size_t)(j0 + it) * LC + lane * 4),
          (v4i_*)&Vlds4[it * 32 + lane], 0, 0);
    }
    if (hi == 0) Alds[l0] = alpha_g[j0 + l0];   // alpha folded at consume time
#else
    // Synchronous fallback: stage alpha[j]*vals[j][0..127] (alpha pre-folded;
    // it is wave-uniform per row -> scalar load + v_mul x4).
    #pragma unroll
    for (int it = 0; it < 16; ++it) {
      const float av = alpha_g[j0 + it];
      float4 v = ((const float4*)(vals_g + (size_t)(j0 + it) * LC))[lane];
      v.x *= av; v.y *= av; v.z *= av; v.w *= av;
      Vlds4[it * 32 + lane] = v;
    }
#endif
    if (jt < 15)                      // pull next tile toward the caches
      __builtin_prefetch(vals_g + (size_t)(j0 + 16) * LC + lane * 16, 0, 0);

    // geometry for this lane's j (both half-waves compute j = j0 + l0)
    const int jj = j0 + l0;
    const float* cj = x + ((size_t)b * N_ + jj) * FIN;
    const float dx = cj[0] - cix, dy = cj[1] - ciy, dz = cj[2] - ciz;
    const float rr = sqrtf(dx * dx + dy * dy + dz * dz + EPSF);
    const float ir = 1.f / rr;
    const float ux = dx * ir, uy = dy * ir, uz = dz * ir;

    if (hi == 0) {
      const float x2 = ux * ux, y2 = uy * uy, z2 = uz * uz;
      float* Y = Ylds + l0 * 16;
      Y[0]  = 0.28209479f;
      Y[1]  = 0.48860251f * uy;  Y[2] = 0.48860251f * uz;  Y[3] = 0.48860251f * ux;
      Y[4]  = 1.09254843f * ux * uy;
      Y[5]  = 1.09254843f * uy * uz;
      Y[6]  = 0.31539157f * (3.f * z2 - 1.f);
      Y[7]  = 1.09254843f * ux * uz;
      Y[8]  = 0.54627422f * (x2 - y2);
      Y[9]  = 0.59004359f * uy * (3.f * x2 - y2);
      Y[10] = 2.89061144f * ux * uy * uz;
      Y[11] = 0.45704579f * uy * (5.f * z2 - 1.f);
      Y[12] = 0.37317633f * uz * (5.f * z2 - 3.f);
      Y[13] = 0.45704579f * ux * (5.f * z2 - 1.f);
      Y[14] = 1.44530572f * uz * (x2 - y2);
      Y[15] = 0.59004359f * ux * (x2 - 3.f * y2);
    }

    // A fragment: h = relu(r*Rw1 + Rb1) computed in packed f16
    const _Float16 rh = (_Float16)rr;
    v16h Af = wh * rh + bh;           // v_pk_fma_f16 x8
    #pragma unroll
    for (int e = 0; e < 16; ++e)
      Af[e] = Af[e] > (_Float16)0.f ? Af[e] : (_Float16)0.f;

#if HAVE_ASYNC_LDS
    __builtin_amdgcn_s_wait_asynccnt(0);  // DMA into LDS complete
#endif
    __syncthreads();                  // LDS tile + Y visible

    // 8 output tiles: R[16 j x 16 lc] = h @ Rw2 tile (WMMA, B from LDS).
    // The fence at the end of each iteration keeps at most one D tile live,
    // holding VGPR pressure far below the 256-register MSB boundary.
    #pragma unroll
    for (int t = 0; t < 8; ++t) {
      const v16h Bf = Bh[t][lane];    // 2x ds_load_b128
      v8f cz = {};
      v8f D = __builtin_amdgcn_wmma_f32_16x16x32_f16(
          false, Af, false, Bf, (short)0, cz, false, false);
      const int nm = mcount[t], yo = moff[t], ao = aoff[t];
      const float rb = rb2v[t];
      #pragma unroll
      for (int r = 0; r < 8; ++r) {
        const int jl = r + hi * 8;    // D row -> local j
#if HAVE_ASYNC_LDS
        const float w = (D[r] + rb) * Alds[jl] * Vlds[jl * LC + t * 16 + l0];
#else
        const float w = (D[r] + rb) * Vlds[jl * LC + t * 16 + l0];
#endif
        const float* Yj = Ylds + jl * 16 + yo;
        #pragma unroll
        for (int m = 0; m < nm; ++m) acc[ao + m] += w * Yj[m];
      }
      asm volatile("" ::: "memory"); // scheduling fence: consume before next tile
    }
  }

  // fold the two half-wave partial sums (rows 0-7 vs 8-15 of every tile)
  #pragma unroll
  for (int m = 0; m < 32; ++m) acc[m] += __shfl_xor(acc[m], 16, 32);

  if (hi == 0) {
    float* inv_g = ws + WS_INV + (size_t)bi * LC;
    #pragma unroll
    for (int t = 0; t < 8; ++t) {
      float s = EPSF;
      #pragma unroll
      for (int m = 0; m < mcount[t]; ++m) {
        const float a = acc[aoff[t] + m];
        s += a * a;
      }
      inv_g[t * 16 + l0] = sqrtf(s);
    }
  }
}

// ---------------------------------------------------------------------------
// Kernel 4: pooling (max/mean over N) + fc1 + layernorm + relu + fc2
// ---------------------------------------------------------------------------
__global__ void __launch_bounds__(256)
head_kernel(const float* __restrict__ fc1_w, const float* __restrict__ fc1_b,
            const float* __restrict__ ln_g, const float* __restrict__ ln_b,
            const float* __restrict__ fc2_w, const float* __restrict__ fc2_b,
            const float* __restrict__ ws_inv, float* __restrict__ out) {
  const int b = blockIdx.x;
  const int t = threadIdx.x;          // 0..255
  __shared__ float pooled[FOUT];
  __shared__ float red[256];
  __shared__ float h1r[FOUT];

  const float* invb = ws_inv + (size_t)b * N_ * LC;
  if (t < LC) {
    float mx = -1e30f;
    for (int i = 0; i < N_; ++i) mx = fmaxf(mx, invb[(size_t)i * LC + t]);
    pooled[t] = mx;
  } else {
    const int f = t - LC;
    float s = 0.f;
    for (int i = 0; i < N_; ++i) s += invb[(size_t)i * LC + f];
    pooled[t] = s * (1.f / N_);
  }
  __syncthreads();

  float h1 = fc1_b[t];
  for (int f = 0; f < FOUT; ++f) h1 += pooled[f] * fc1_w[f * FOUT + t];

  red[t] = h1; __syncthreads();
  for (int s = 128; s > 0; s >>= 1) { if (t < s) red[t] += red[t + s]; __syncthreads(); }
  const float mu = red[0] * (1.f / FOUT); __syncthreads();
  const float dv = h1 - mu;
  red[t] = dv * dv; __syncthreads();
  for (int s = 128; s > 0; s >>= 1) { if (t < s) red[t] += red[t + s]; __syncthreads(); }
  const float var = red[0] * (1.f / FOUT);

  const float hn = dv * rsqrtf(var + 1e-5f) * ln_g[t] + ln_b[t];
  h1r[t] = fmaxf(hn, 0.f);
  __syncthreads();

  if (t < DOUT) {
    float o = fc2_b[t];
    for (int f = 0; f < FOUT; ++f) o += h1r[f] * fc2_w[f * DOUT + t];
    out[b * DOUT + t] = o;
  }
}

// ---------------------------------------------------------------------------
extern "C" void kernel_launch(void* const* d_in, const int* in_sizes, int n_in,
                              void* d_out, int out_size, void* d_ws, size_t ws_size,
                              hipStream_t stream) {
  const float* x    = (const float*)d_in[0];
  const float* Wq   = (const float*)d_in[1];
  const float* Wk   = (const float*)d_in[2];
  const float* Wv   = (const float*)d_in[3];
  const float* Rw1  = (const float*)d_in[4];
  const float* Rb1  = (const float*)d_in[5];
  const float* Rw2  = (const float*)d_in[6];
  const float* Rb2  = (const float*)d_in[7];
  const float* fc1w = (const float*)d_in[8];
  const float* fc1b = (const float*)d_in[9];
  const float* lng  = (const float*)d_in[10];
  const float* lnb  = (const float*)d_in[11];
  const float* fc2w = (const float*)d_in[12];
  const float* fc2b = (const float*)d_in[13];
  float* ws  = (float*)d_ws;
  float* out = (float*)d_out;

  prep_kernel<<<B_ * N_, 128, 0, stream>>>(x, Wq, Wk, Wv, ws);
  attn_kernel<<<B_ * N_, 256, 0, stream>>>(ws);
  pair_kernel<<<B_ * N_, 32, 0, stream>>>(x, Rw1, Rb1, Rw2, Rb2, ws);
  head_kernel<<<B_, 256, 0, stream>>>(fc1w, fc1b, lng, lnb, fc2w, fc2b,
                                      ws + WS_INV, out);
}
